// Transformer_3496103379194
// MI455X (gfx1250) — matrix operand order repaired and checked by reading, not compiled, not run on hardware
//
#include <hip/hip_runtime.h>
#include <hip/hip_bf16.h>
#include <math.h>

// ---------------- model constants (match reference) ----------------
#define N_LAYERS 6
#define D_MODEL  512
#define NHEAD    8
#define DKH      64
#define DVH      64
#define DFF      2048
#define VOCAB    32000
#define BATCH    4
#define SEQ      512
#define MTOK     (BATCH*SEQ)   // 2048 tokens

typedef __attribute__((ext_vector_type(16))) __bf16 v16bf;
typedef __attribute__((ext_vector_type(8)))  __bf16 v8bf;
typedef __attribute__((ext_vector_type(4)))  __bf16 v4bf;
typedef __attribute__((ext_vector_type(8)))  float  v8f;
typedef __attribute__((ext_vector_type(4)))  float  v4f;

#define APITCH 36   // fp32 elements per A-tile LDS row (32 + 4 pad)
#define BPITCH 40   // bf16 elements per B-tile LDS row (32 + 8 pad)

// =====================================================================
// Generic batched GEMM: C = A(MxK,f32) * B(KxN,f32) [+bias] [relu]
// bf16 WMMA 16x16x32, f32 accumulate. Block tile 128(M) x 64(N), K-step 32.
// 256 threads = 8 waves arranged 2(M) x 4(N); each wave owns 64x16 of C.
//   - A tile: fp32 staged straight into LDS with global_load_async_to_lds_b128
//     (ASYNCcnt), converted to bf16 during fragment assembly.
//   - B tile: strided-addressed B(k,n)=B[k*ldbK + n*ldbN]; float4 fast paths
//     for ldbK==1 / ldbN==1, staged via registers into bf16 LDS [n][k].
//   - Double-buffered LDS, one barrier per K-step.
//   - All fragments assembled into distinct registers BEFORE the 4 WMMAs to
//     avoid per-WMMA WAR hazard NOPs (ISA 7.12.1).
// Batch z -> (zo=z/inner, zi=z%inner) with independent strides per operand.
// =====================================================================
__global__ __launch_bounds__(256) void gemm_bf16_wmma(
    const float* __restrict__ A, const float* __restrict__ B,
    float* __restrict__ C, const float* __restrict__ bias,
    int M, int N, int K,
    int lda, int ldbK, int ldbN, int ldc,
    long long sAo, long long sAi, long long sBo, long long sBi,
    long long sCo, long long sCi, int innerCount, int relu)
{
    __shared__ alignas(16) float  Asf[2][128][APITCH]; // [buf][m][k] fp32
    __shared__ alignas(16) __bf16 Bsf[2][64][BPITCH];  // [buf][n][k] bf16

    const int z  = blockIdx.z;
    const int zo = z / innerCount, zi = z % innerCount;
    A += zo * sAo + zi * sAi;
    B += zo * sBo + zi * sBi;
    C += zo * sCo + zi * sCi;

    const int m0   = blockIdx.y * 128;
    const int n0   = blockIdx.x * 64;
    const int tid  = threadIdx.x;
    const int lane = tid & 31;
    const int wave = tid >> 5;
    const int wm   = (wave >> 2) * 64;   // wave M offset within block tile
    const int wn   = (wave & 3) * 16;    // wave N offset within block tile
    const int half = lane >> 4;          // lane group (0: lanes 0-15, 1: 16-31)
    const int l16  = lane & 15;

    const bool fullM = (m0 + 128 <= M);
    const bool fullN = (n0 + 64 <= N);
    const bool bPathK = (ldbK == 1) && fullN;  // K-contiguous B (e.g. K^T)
    const bool bPathN = (ldbN == 1) && fullN;  // N-contiguous B (weights)

    v8f acc[4] = {};
    float br[8];      // B staging registers (8 fp32 per thread per K-step)

    // ---------------- staging helpers ----------------
    auto stageA = [&](int k0, int buf) {
        if (fullM && (k0 + 32 <= K)) {
            // 128x32 fp32 tile = 1024 x 16B chunks; 4 async DMAs per thread
            #pragma unroll
            for (int i = 0; i < 4; ++i) {
                int e = tid + i * 256;
                int r = e >> 3, c4 = (e & 7) * 4;
                const float* gp = A + (long long)(m0 + r) * lda + (k0 + c4);
                unsigned loff = (unsigned)(size_t)(const void*)&Asf[buf][r][c4];
                asm volatile("global_load_async_to_lds_b128 %0, %1, off"
                             :: "v"(loff), "v"(gp) : "memory");
            }
        } else {
            // guarded scalar fallback (never taken for this model's shapes)
            #pragma unroll
            for (int i = 0; i < 16; ++i) {
                int e = tid + i * 256;
                int r = e >> 5, c = e & 31;
                int gm = m0 + r, gk = k0 + c;
                Asf[buf][r][c] = (gm < M && gk < K)
                                   ? A[(long long)gm * lda + gk] : 0.0f;
            }
        }
    };
    auto loadB = [&](int k0) {
        if (bPathK && (k0 + 32 <= K)) {
            #pragma unroll
            for (int i = 0; i < 2; ++i) {
                int e = tid + i * 256;
                int n = e >> 3, c4 = (e & 7) * 4;
                v4f t = *(const v4f*)(B + (long long)(n0 + n) * ldbN + (k0 + c4));
                #pragma unroll
                for (int j = 0; j < 4; ++j) br[i * 4 + j] = t[j];
            }
        } else if (bPathN && (k0 + 32 <= K)) {
            #pragma unroll
            for (int i = 0; i < 2; ++i) {
                int e = tid + i * 256;
                int k = e >> 4, n4 = (e & 15) * 4;
                v4f t = *(const v4f*)(B + (long long)(k0 + k) * ldbK + (n0 + n4));
                #pragma unroll
                for (int j = 0; j < 4; ++j) br[i * 4 + j] = t[j];
            }
        } else {
            #pragma unroll
            for (int i = 0; i < 8; ++i) {
                int e = tid + i * 256;
                int n = e >> 5, c = e & 31;
                int gn = n0 + n, gk = k0 + c;
                br[i] = (gn < N && gk < K)
                          ? B[(long long)gk * ldbK + (long long)gn * ldbN] : 0.0f;
            }
        }
    };
    auto commitB = [&](int k0, int buf) {
        if (bPathK && (k0 + 32 <= K)) {
            #pragma unroll
            for (int i = 0; i < 2; ++i) {
                int e = tid + i * 256;
                int n = e >> 3, c4 = (e & 7) * 4;
                v4bf o;
                #pragma unroll
                for (int j = 0; j < 4; ++j) o[j] = (__bf16)br[i * 4 + j];
                *(v4bf*)&Bsf[buf][n][c4] = o;
            }
        } else if (bPathN && (k0 + 32 <= K)) {
            #pragma unroll
            for (int i = 0; i < 2; ++i) {
                int e = tid + i * 256;
                int k = e >> 4, n4 = (e & 15) * 4;
                #pragma unroll
                for (int j = 0; j < 4; ++j)
                    Bsf[buf][n4 + j][k] = (__bf16)br[i * 4 + j];
            }
        } else {
            #pragma unroll
            for (int i = 0; i < 8; ++i) {
                int e = tid + i * 256;
                int n = e >> 5, c = e & 31;
                Bsf[buf][n][c] = (__bf16)br[i];
            }
        }
    };

    // ---------------- prologue: stage tile 0 ----------------
    stageA(0, 0);
    loadB(0);
    commitB(0, 0);
    asm volatile("s_wait_asynccnt 0x0" ::: "memory");
    __syncthreads();

    int cur = 0;
    for (int k0 = 0; k0 < K; k0 += 32) {
        const bool hasNext = (k0 + 32) < K;
        if (hasNext) {            // kick off next tile's global traffic early
            stageA(k0 + 32, cur ^ 1);
            loadB(k0 + 32);
        }

        // ---- B fragment: lane -> N = l16, elems 0-7: Kloc = half*8+j ; elems 8-15: Kloc = 16+half*8+j ----
        v8bf b_lo = *(const v8bf*)&Bsf[cur][wn + l16][half * 8];
        v8bf b_hi = *(const v8bf*)&Bsf[cur][wn + l16][16 + half * 8];
        v16bf bfrag = __builtin_shufflevector(b_lo, b_hi,
            0,1,2,3,4,5,6,7,8,9,10,11,12,13,14,15);

        // ---- assemble ALL 4 A fragments first (distinct regs), then WMMAs
        //      back-to-back: avoids WMMA->VALU WAR hazard NOPs per WMMA ----
        // elems 0-7: Kloc = half*8+j ; elems 8-15: Kloc = 16+half*8+j
        v16bf afrag[4];
        #pragma unroll
        for (int t = 0; t < 4; ++t) {
            const float* ar = &Asf[cur][wm + t * 16 + l16][0];
            v8f f0 = *(const v8f*)(ar + half * 8);
            v8f f1 = *(const v8f*)(ar + 16 + half * 8);
            #pragma unroll
            for (int j = 0; j < 8; ++j) {
                afrag[t][j]     = (__bf16)f0[j];
                afrag[t][j + 8] = (__bf16)f1[j];
            }
        }
        #pragma unroll
        for (int t = 0; t < 4; ++t) {
            acc[t] = __builtin_amdgcn_wmma_f32_16x16x32_bf16(
                false, afrag[t], false, bfrag, (short)0, acc[t], false, false);
        }

        if (hasNext) {
            commitB(k0 + 32, cur ^ 1);
            asm volatile("s_wait_asynccnt 0x0" ::: "memory");
        }
        __syncthreads();
        cur ^= 1;
    }

    // ---- epilogue: D layout lane -> N = l16, VGPR i -> M = half*8 + i ----
    #pragma unroll
    for (int t = 0; t < 4; ++t) {
        #pragma unroll
        for (int i = 0; i < 8; ++i) {
            int gm = m0 + wm + t * 16 + half * 8 + i;
            int gn = n0 + wn + l16;
            if (gm < M && gn < N) {
                float v = acc[t][i];
                if (bias) v += bias[gn];
                if (relu) v = fmaxf(v, 0.0f);
                C[(long long)gm * ldc + gn] = v;
            }
        }
    }
}

// =====================================================================
// Embedding lookup + (buggy-vs-paper) positional encoding, fp32 out.
// =====================================================================
__global__ __launch_bounds__(256) void embed_pe_kernel(
    const int* __restrict__ tok, const float* __restrict__ embed,
    float* __restrict__ out)
{
    int ti = blockIdx.x;            // b*SEQ + s
    int s  = ti % SEQ;
    long long t = (long long)tok[ti];
    for (int d = threadIdx.x; d < D_MODEL; d += 256) {
        float angle = (float)s / __powf(10000.0f, 2.0f * (float)d / (float)D_MODEL);
        float pe = ((d & 1) == 0) ? __sinf(angle) : __cosf(angle);
        out[(long long)ti * D_MODEL + d] = embed[t * D_MODEL + d] + pe;
    }
}

// =====================================================================
// Masked scaled softmax over last dim of scores (B,H,S,T), in place.
// grid = (S, H, B), block = 256.
// =====================================================================
__global__ __launch_bounds__(256) void softmax_mask_kernel(
    float* __restrict__ scores,
    const unsigned char* __restrict__ qmask,
    const unsigned char* __restrict__ kmask,
    int S, int T, int causal, float scale)
{
    int s = blockIdx.x, h = blockIdx.y, b = blockIdx.z;
    int H = gridDim.y;
    float* row = scores + (((long long)(b * H + h) * S + s) * (long long)T);
    __shared__ float red[256];
    int tid = threadIdx.x;
    bool qok = qmask[(long long)b * S + s] != 0;

    float lmax = -INFINITY;
    for (int t = tid; t < T; t += 256) {
        bool ok = qok && (kmask[(long long)b * T + t] != 0) && (!causal || t <= s);
        float v = ok ? row[t] * scale : -INFINITY;
        row[t] = v;
        lmax = fmaxf(lmax, v);
    }
    red[tid] = lmax; __syncthreads();
    for (int off = 128; off > 0; off >>= 1) {
        if (tid < off) red[tid] = fmaxf(red[tid], red[tid + off]);
        __syncthreads();
    }
    float m = red[0]; __syncthreads();

    float lsum = 0.0f;
    for (int t = tid; t < T; t += 256) {
        float e = __expf(row[t] - m);
        row[t] = e;
        lsum += e;
    }
    red[tid] = lsum; __syncthreads();
    for (int off = 128; off > 0; off >>= 1) {
        if (tid < off) red[tid] += red[tid + off];
        __syncthreads();
    }
    float inv = 1.0f / red[0];
    __syncthreads();
    for (int t = tid; t < T; t += 256) row[t] *= inv;
}

// =====================================================================
// x = LayerNorm(x + a) * g + b   (in place on x). One block per row.
// =====================================================================
__global__ __launch_bounds__(256) void add_ln_kernel(
    float* __restrict__ x, const float* __restrict__ a,
    const float* __restrict__ g, const float* __restrict__ beta)
{
    long long r = blockIdx.x;
    float* row = x + r * D_MODEL;
    const float* ar = a + r * D_MODEL;
    __shared__ float red[256];
    int tid = threadIdx.x;
    int d0 = tid, d1 = tid + 256;

    float v0 = row[d0] + ar[d0];
    float v1 = row[d1] + ar[d1];

    red[tid] = v0 + v1; __syncthreads();
    for (int off = 128; off > 0; off >>= 1) {
        if (tid < off) red[tid] += red[tid + off];
        __syncthreads();
    }
    float mu = red[0] * (1.0f / D_MODEL); __syncthreads();

    float e0 = v0 - mu, e1 = v1 - mu;
    red[tid] = e0 * e0 + e1 * e1; __syncthreads();
    for (int off = 128; off > 0; off >>= 1) {
        if (tid < off) red[tid] += red[tid + off];
        __syncthreads();
    }
    float rstd = rsqrtf(red[0] * (1.0f / D_MODEL) + 1e-5f);

    row[d0] = e0 * rstd * g[d0] + beta[d0];
    row[d1] = e1 * rstd * g[d1] + beta[d1];
}

// =====================================================================
// Host orchestration
// =====================================================================
extern "C" void kernel_launch(void* const* d_in, const int* in_sizes, int n_in,
                              void* d_out, int out_size, void* d_ws, size_t ws_size,
                              hipStream_t stream)
{
    (void)in_sizes; (void)n_in; (void)out_size; (void)ws_size;

    const int*           src_tok  = (const int*)d_in[0];
    const int*           tgt_tok  = (const int*)d_in[1];
    const unsigned char* src_mask = (const unsigned char*)d_in[2];
    const unsigned char* tgt_mask = (const unsigned char*)d_in[3];
    const float* embed     = (const float*)d_in[4];
    const float* enc_wqkv  = (const float*)d_in[5];
    const float* enc_wo    = (const float*)d_in[6];
    const float* enc_ff_w1 = (const float*)d_in[7];
    const float* enc_ff_b1 = (const float*)d_in[8];
    const float* enc_ff_w2 = (const float*)d_in[9];
    const float* enc_ff_b2 = (const float*)d_in[10];
    const float* enc_ln_g  = (const float*)d_in[11];
    const float* enc_ln_b  = (const float*)d_in[12];
    const float* dec_wqkv1 = (const float*)d_in[13];
    const float* dec_wo1   = (const float*)d_in[14];
    const float* dec_wqkv2 = (const float*)d_in[15];
    const float* dec_wo2   = (const float*)d_in[16];
    const float* dec_ff_w1 = (const float*)d_in[17];
    const float* dec_ff_b1 = (const float*)d_in[18];
    const float* dec_ff_w2 = (const float*)d_in[19];
    const float* dec_ff_b2 = (const float*)d_in[20];
    const float* dec_ln_g  = (const float*)d_in[21];
    const float* dec_ln_b  = (const float*)d_in[22];
    const float* out_w     = (const float*)d_in[23];
    const float* out_b     = (const float*)d_in[24];

    // ---- workspace layout (floats) ----
    float* ws = (float*)d_ws;
    float* x       = ws;                                   // (B,S,D)
    float* y       = x + (long long)MTOK * D_MODEL;
    float* q       = y + (long long)MTOK * D_MODEL;        // (B,S,H,DK)
    float* kbuf    = q + (long long)MTOK * D_MODEL;
    float* vbuf    = kbuf + (long long)MTOK * D_MODEL;
    float* attnout = vbuf + (long long)MTOK * D_MODEL;     // (B,S,H,DV)
    float* tmp     = attnout + (long long)MTOK * D_MODEL;  // (B,S,D)
    float* ffmid   = tmp + (long long)MTOK * D_MODEL;      // (B,S,DFF)
    float* scores  = ffmid + (long long)MTOK * DFF;        // (B,H,S,S)

    auto gemm = [&](const float* A, const float* B, float* C, const float* bias,
                    int M, int N, int K, int lda, int ldbK, int ldbN, int ldc,
                    long long sAo, long long sAi, long long sBo, long long sBi,
                    long long sCo, long long sCi, int inner, int batches, int relu) {
        dim3 grid((N + 63) / 64, (M + 127) / 128, batches);
        gemm_bf16_wmma<<<grid, 256, 0, stream>>>(A, B, C, bias, M, N, K,
            lda, ldbK, ldbN, ldc, sAo, sAi, sBo, sBi, sCo, sCi, inner, relu);
    };

    const long long HD  = (long long)NHEAD * DKH;           // 512
    const long long WQK = (long long)NHEAD * D_MODEL * DKH; // one of wq/wk/wv

    auto attention = [&](const float* q_in, const float* kv_in,
                         const float* wqkv /* (3,H,D,DK) */, const float* wo,
                         const unsigned char* qm, const unsigned char* km,
                         int causal, float* out) {
        const float* wq = wqkv;
        const float* wk = wqkv + WQK;
        const float* wv = wqkv + 2 * WQK;
        // Q/K/V projections, batched over heads (A shared across batches)
        gemm(q_in, wq, q, nullptr, MTOK, DKH, D_MODEL, D_MODEL, DKH, 1, (int)HD,
             0, 0, 0, (long long)D_MODEL * DKH, 0, DKH, NHEAD, NHEAD, 0);
        gemm(kv_in, wk, kbuf, nullptr, MTOK, DKH, D_MODEL, D_MODEL, DKH, 1, (int)HD,
             0, 0, 0, (long long)D_MODEL * DKH, 0, DKH, NHEAD, NHEAD, 0);
        gemm(kv_in, wv, vbuf, nullptr, MTOK, DKH, D_MODEL, D_MODEL, DKH, 1, (int)HD,
             0, 0, 0, (long long)D_MODEL * DKH, 0, DKH, NHEAD, NHEAD, 0);
        // scores(b,h) = Q(b,h) [SxDK] * K(b,h)^T   (B with ldbK==1 fast path)
        gemm(q, kbuf, scores, nullptr, SEQ, SEQ, DKH,
             (int)HD, 1, (int)HD, SEQ,
             (long long)SEQ * HD, DKH, (long long)SEQ * HD, DKH,
             (long long)NHEAD * SEQ * SEQ, (long long)SEQ * SEQ,
             NHEAD, BATCH * NHEAD, 0);
        softmax_mask_kernel<<<dim3(SEQ, NHEAD, BATCH), 256, 0, stream>>>(
            scores, qm, km, SEQ, SEQ, causal, rsqrtf((float)DKH));
        // attnout(b,h) = P [SxS] * V(b,h) [SxDV]
        gemm(scores, vbuf, attnout, nullptr, SEQ, DVH, SEQ,
             SEQ, (int)HD, 1, (int)HD,
             (long long)NHEAD * SEQ * SEQ, (long long)SEQ * SEQ,
             (long long)SEQ * HD, DVH,
             (long long)SEQ * HD, DVH,
             NHEAD, BATCH * NHEAD, 0);
        // out = attnout [MTOK x H*DV] * wo [H*DV x D]
        gemm(attnout, wo, out, nullptr, MTOK, D_MODEL, (int)HD,
             (int)HD, D_MODEL, 1, D_MODEL, 0, 0, 0, 0, 0, 0, 1, 1, 0);
    };

    auto ffn = [&](float* io, const float* w1, const float* b1,
                   const float* w2, const float* b2) {
        gemm(io, w1, ffmid, b1, MTOK, DFF, D_MODEL, D_MODEL, DFF, 1, DFF,
             0, 0, 0, 0, 0, 0, 1, 1, /*relu=*/1);
        gemm(ffmid, w2, tmp, b2, MTOK, D_MODEL, DFF, DFF, D_MODEL, 1, D_MODEL,
             0, 0, 0, 0, 0, 0, 1, 1, 0);
    };

    auto add_ln = [&](float* io, const float* a, const float* g, const float* b) {
        add_ln_kernel<<<MTOK, 256, 0, stream>>>(io, a, g, b);
    };

    // ---------------- encoder ----------------
    embed_pe_kernel<<<MTOK, 256, 0, stream>>>(src_tok, embed, x);
    for (int i = 0; i < N_LAYERS; ++i) {
        const float* wqkv = enc_wqkv + (long long)i * 3 * WQK;
        attention(x, x, wqkv, enc_wo + (long long)i * HD * D_MODEL,
                  src_mask, src_mask, /*causal=*/0, tmp);
        add_ln(x, tmp, enc_ln_g + (long long)(i * 2 + 0) * D_MODEL,
                       enc_ln_b + (long long)(i * 2 + 0) * D_MODEL);
        ffn(x, enc_ff_w1 + (long long)i * D_MODEL * DFF,
               enc_ff_b1 + (long long)i * DFF,
               enc_ff_w2 + (long long)i * DFF * D_MODEL,
               enc_ff_b2 + (long long)i * D_MODEL);
        add_ln(x, tmp, enc_ln_g + (long long)(i * 2 + 1) * D_MODEL,
                       enc_ln_b + (long long)(i * 2 + 1) * D_MODEL);
    }

    // ---------------- decoder ----------------
    embed_pe_kernel<<<MTOK, 256, 0, stream>>>(tgt_tok, embed, y);
    for (int i = 0; i < N_LAYERS; ++i) {
        const float* wqkv1 = dec_wqkv1 + (long long)i * 3 * WQK;
        attention(y, y, wqkv1, dec_wo1 + (long long)i * HD * D_MODEL,
                  tgt_mask, tgt_mask, /*causal=*/1, tmp);
        add_ln(y, tmp, dec_ln_g + (long long)(i * 3 + 0) * D_MODEL,
                       dec_ln_b + (long long)(i * 3 + 0) * D_MODEL);
        const float* wqkv2 = dec_wqkv2 + (long long)i * 3 * WQK;
        attention(y, x, wqkv2, dec_wo2 + (long long)i * HD * D_MODEL,
                  tgt_mask, src_mask, /*causal=*/0, tmp);
        add_ln(y, tmp, dec_ln_g + (long long)(i * 3 + 1) * D_MODEL,
                       dec_ln_b + (long long)(i * 3 + 1) * D_MODEL);
        ffn(y, dec_ff_w1 + (long long)i * D_MODEL * DFF,
               dec_ff_b1 + (long long)i * DFF,
               dec_ff_w2 + (long long)i * DFF * D_MODEL,
               dec_ff_b2 + (long long)i * D_MODEL);
        add_ln(y, tmp, dec_ln_g + (long long)(i * 3 + 2) * D_MODEL,
                       dec_ln_b + (long long)(i * 3 + 2) * D_MODEL);
    }

    // ---------------- final projection to vocab (fp32 out) ----------------
    gemm(y, out_w, (float*)d_out, out_b, MTOK, VOCAB, D_MODEL,
         D_MODEL, VOCAB, 1, VOCAB, 0, 0, 0, 0, 0, 0, 1, 1, 0);
}
